// GRU_Decoder_75582834475298
// MI455X (gfx1250) — matrix operand order repaired and checked by reading, not compiled, not run on hardware
//
#include <hip/hip_runtime.h>
#include <hip/hip_bf16.h>

#define NL  8
#define H   1024
#define EMB 1024
#define B   256
#define T   96
#define EPS 1e-6f

#define GRID_BLOCKS 128   // 16 mtiles x 8 ntile-groups, 8 waves/block -> 1024 tiles

typedef __bf16 bf16_t;
typedef __attribute__((ext_vector_type(8)))  bf16_t v8bf;
typedef __attribute__((ext_vector_type(16))) bf16_t v16bf;
typedef __attribute__((ext_vector_type(8)))  float  v8f;

static __device__ __forceinline__ v16bf cat8(v8bf lo, v8bf hi) {
  return __builtin_shufflevector(lo, hi, 0,1,2,3,4,5,6,7,8,9,10,11,12,13,14,15);
}
static __device__ __forceinline__ float frcp(float x) {
  return __builtin_amdgcn_rcpf(x);            // single v_rcp_f32, no IEEE div expansion
}
static __device__ __forceinline__ float sigm(float x) {
  return frcp(1.0f + __expf(-x));
}
static __device__ __forceinline__ float tanh_f(float x) {
  float e = __expf(-2.0f * fabsf(x));
  return copysignf((1.0f - e) * frcp(1.0f + e), x);
}
static __device__ __forceinline__ v8f wmma_bf16(v16bf a, v16bf b, v8f c) {
  return __builtin_amdgcn_wmma_f32_16x16x32_bf16(false, a, false, b, (short)0, c, false, false);
}

// ---------------------------------------------------------------------------
// Device-wide sense barrier (all GRID_BLOCKS co-resident: 128 blocks x 8 waves
// is far below MI455X WGP capacity). Release/acquire at agent scope so stores
// drain to L2 and WGP caches are invalidated across phases.
// ---------------------------------------------------------------------------
struct GridBar { int count; int phase; };

static __device__ __forceinline__ void grid_barrier(GridBar* bar, int& phase) {
  __threadfence();                 // release: drain my stores to device scope
  __syncthreads();
  if (threadIdx.x == 0) {
    const int p = phase + 1;
    int ticket = __hip_atomic_fetch_add(&bar->count, 1, __ATOMIC_ACQ_REL,
                                        __HIP_MEMORY_SCOPE_AGENT);
    if (ticket == GRID_BLOCKS - 1) {
      __hip_atomic_store(&bar->count, 0, __ATOMIC_RELAXED, __HIP_MEMORY_SCOPE_AGENT);
      __hip_atomic_store(&bar->phase, p, __ATOMIC_RELEASE, __HIP_MEMORY_SCOPE_AGENT);
    } else {
      while (__hip_atomic_load(&bar->phase, __ATOMIC_ACQUIRE,
                               __HIP_MEMORY_SCOPE_AGENT) < p) {
        __builtin_amdgcn_s_sleep(2);
      }
    }
  }
  __syncthreads();
  __threadfence();                 // acquire: invalidate stale WGP-cache lines
  ++phase;
}

// ---------------------------------------------------------------------------
// Weight conversion: fp32 [N,K] row-major -> bf16 B-fragment blobs.
// Blob per (kb, ntile): 512 bf16, element = lane*16 + i with
//   N = ntile*16 + (lane & 15)
//   K = kb*32 + (lane < 16 ? 0 : 8) + (i < 8 ? i : i + 8)   (ISA 16-bit B layout)
// GRU matrices: [32 kb][192 nt][512]; W_lin: [64 kb][64 nt][512].
// ---------------------------------------------------------------------------
__global__ __launch_bounds__(256)
void conv_weights_kernel(const float* __restrict__ Wih, const float* __restrict__ Whh,
                         const float* __restrict__ Wlin,
                         bf16_t* __restrict__ WiB, bf16_t* __restrict__ WhB,
                         bf16_t* __restrict__ WlB) {
  const size_t MW  = (size_t)3 * H * H;
  const size_t NWI = (size_t)NL * MW;
  size_t id = (size_t)blockIdx.x * 256 + threadIdx.x;

  const float* src; bf16_t* dst; size_t within; bool gru;
  if (id < NWI)          { src = Wih;  dst = WiB; within = id;           gru = true;  }
  else if (id < 2 * NWI) { src = Whh;  dst = WhB; within = id - NWI;     gru = true;  }
  else                   { src = Wlin; dst = WlB; within = id - 2 * NWI; gru = false; }

  if (gru) {
    size_t l  = within / MW;
    size_t r  = within - l * MW;
    size_t kb = r / (192u * 512u);
    size_t r2 = r - kb * (192u * 512u);
    size_t nt = r2 >> 9;
    size_t e  = r2 & 511u;
    int lane = (int)(e >> 4), i = (int)(e & 15);
    int n = (int)(nt << 4) + (lane & 15);
    int k = (int)(kb << 5) + ((lane < 16) ? 0 : 8) + (i < 8 ? i : i + 8);
    dst[within] = (bf16_t)src[l * MW + (size_t)n * H + k];
  } else {
    size_t kb = within / (64u * 512u);
    size_t r2 = within - kb * (64u * 512u);
    size_t nt = r2 >> 9;
    size_t e  = r2 & 511u;
    int lane = (int)(e >> 4), i = (int)(e & 15);
    int n = (int)(nt << 4) + (lane & 15);
    int k = (int)(kb << 5) + ((lane < 16) ? 0 : 8) + (i < 8 ? i : i + 8);
    dst[within] = (bf16_t)src[(size_t)n * (2 * H) + k];
  }
}

__global__ __launch_bounds__(256)
void conv_state_kernel(const float* __restrict__ G, float* __restrict__ hf,
                       bf16_t* __restrict__ hb, GridBar* bar) {
  size_t id = (size_t)blockIdx.x * 256 + threadIdx.x;   // NL*B*H threads
  float v = G[id];
  hf[id] = v;
  hb[id] = (bf16_t)v;
  if (id == 0) { bar->count = 0; bar->phase = 0; }      // re-arm barrier each call
}

// ---------------------------------------------------------------------------
// K-loop for one GRU cell tile: 6 x v_wmma_f32_16x16x32_bf16 per 32-wide
// K-step. acc_r/acc_z fuse x- and h-GEMMs; i_n / h_n kept separate.
// ---------------------------------------------------------------------------
template <bool AF32>
static __device__ __forceinline__ void gru_kloop(
    const float* Af, const bf16_t* Ab, int lda, const bf16_t* Hbp,
    const bf16_t* Wi, const bf16_t* Wh,
    int row, int khalf, size_t lofs, int ntile,
    v8f& ar, v8f& az, v8f& ain, v8f& ahn) {
  #pragma unroll 2
  for (int kb = 0; kb < 32; ++kb) {
    const int k0 = (kb << 5) + khalf;

    v16bf avx;
    if (AF32) {
      v8f f0 = *(const v8f*)(Af + (size_t)row * lda + k0);
      v8f f1 = *(const v8f*)(Af + (size_t)row * lda + k0 + 16);
      v16bf tv;
      #pragma unroll
      for (int i = 0; i < 8; ++i) { tv[i] = (bf16_t)f0[i]; tv[i + 8] = (bf16_t)f1[i]; }
      avx = tv;
    } else {
      v8bf lo = *(const v8bf*)(Ab + (size_t)row * lda + k0);
      v8bf hi = *(const v8bf*)(Ab + (size_t)row * lda + k0 + 16);
      avx = cat8(lo, hi);
    }
    v8bf hlo = *(const v8bf*)(Hbp + (size_t)row * H + k0);
    v8bf hhi = *(const v8bf*)(Hbp + (size_t)row * H + k0 + 16);
    v16bf avh = cat8(hlo, hhi);

    const bf16_t* wib = Wi + (((size_t)kb * 192 + ntile) << 9) + lofs;
    const bf16_t* whb = Wh + (((size_t)kb * 192 + ntile) << 9) + lofs;
    v16bf bri = *(const v16bf*)(wib);
    v16bf bzi = *(const v16bf*)(wib + ((size_t)64 << 9));
    v16bf bni = *(const v16bf*)(wib + ((size_t)128 << 9));
    v16bf brh = *(const v16bf*)(whb);
    v16bf bzh = *(const v16bf*)(whb + ((size_t)64 << 9));
    v16bf bnh = *(const v16bf*)(whb + ((size_t)128 << 9));
    if (kb < 31) {
      __builtin_prefetch(wib + ((size_t)192 << 9), 0, 1);
      __builtin_prefetch(whb + ((size_t)192 << 9), 0, 1);
    }

    ar  = wmma_bf16(avx, bri, ar);
    ar  = wmma_bf16(avh, brh, ar);
    az  = wmma_bf16(avx, bzi, az);
    az  = wmma_bf16(avh, bzh, az);
    ain = wmma_bf16(avx, bni, ain);
    ahn = wmma_bf16(avh, bnh, ahn);
  }
}

// ---------------------------------------------------------------------------
// Persistent fused decoder: entire T x (8 GRU layers + lin) recurrence in ONE
// kernel; device-wide barrier between layers. 4 launches total per call.
// Tile map: wave -> 16x16 output tile. mtile = bid&15, ntile = (bid>>4)*8+wid.
// ---------------------------------------------------------------------------
__global__ __launch_bounds__(256)
void fused_decoder_kernel(const float* __restrict__ X,
                          const bf16_t* __restrict__ WiB,
                          const bf16_t* __restrict__ WhB,
                          const bf16_t* __restrict__ WlB,
                          float* hAf, float* hBf, bf16_t* hAb, bf16_t* hBb,
                          float* __restrict__ out, GridBar* bar) {
  const int wid   = threadIdx.x >> 5;
  const int lane  = threadIdx.x & 31;
  const int mtile = blockIdx.x & 15;
  const int ntile = ((blockIdx.x >> 4) << 3) + wid;
  const int row   = (mtile << 4) + (lane & 15);
  const int khalf = (lane < 16) ? 0 : 8;
  const size_t lofs = (size_t)lane << 4;
  const int n     = (ntile << 4) + (lane & 15);
  const int mbase = (mtile << 4) + ((lane < 16) ? 0 : 8);
  const size_t S  = (size_t)B * H;

  float*  sAf = hAf; float*  sBf = hBf;
  bf16_t* sAb = hAb; bf16_t* sBb = hBb;
  int phase = 0;

  for (int t = 0; t < T; ++t) {
    // ---- 8 stacked GRU cells -------------------------------------------
    for (int l = 0; l < NL; ++l) {
      const bf16_t* Wi  = WiB + (size_t)l * 3 * H * H;
      const bf16_t* Wh  = WhB + (size_t)l * 3 * H * H;
      const bf16_t* Hbp = sAb + (size_t)l * S;
      const float*  Hfp = sAf + (size_t)l * S;
      float*  Of = sBf + (size_t)l * S;
      bf16_t* Ob = sBb + (size_t)l * S;

      v8f ar = {}, az = {}, ain = {}, ahn = {};
      if (l == 0) {
        gru_kloop<true>(X + (size_t)t * EMB, nullptr, T * EMB, Hbp, Wi, Wh,
                        row, khalf, lofs, ntile, ar, az, ain, ahn);
      } else {
        gru_kloop<false>(nullptr, sBb + (size_t)(l - 1) * S, H, Hbp, Wi, Wh,
                         row, khalf, lofs, ntile, ar, az, ain, ahn);
      }
      // pointwise: C/D layout -> VGPR g, lane: M = g + (lane>=16 ? 8:0), N = lane&15
      #pragma unroll
      for (int g = 0; g < 8; ++g) {
        const int m = mbase + g;
        float r  = sigm(ar[g]);
        float z  = sigm(az[g]);
        float nn = tanh_f(ain[g] + r * ahn[g]);
        float hp = Hfp[(size_t)m * H + n];
        float ho = (1.0f - z) * nn + z * hp;
        Of[(size_t)m * H + n] = ho;
        Ob[(size_t)m * H + n] = (bf16_t)ho;
      }
      grid_barrier(bar, phase);
    }

    // ---- h1 = tanh([h0_old, hs7] @ Wlin^T) -> new h0 + output row ------
    {
      const bf16_t* A0 = sAb;                     // old h0
      const bf16_t* A1 = sBb + (size_t)7 * S;     // hs7
      v8f acc = {};
      #pragma unroll 2
      for (int kb = 0; kb < 64; ++kb) {
        const bf16_t* src = (kb < 32) ? A0 : A1;
        const int k0 = ((kb & 31) << 5) + khalf;
        v8bf lo = *(const v8bf*)(src + (size_t)row * H + k0);
        v8bf hi = *(const v8bf*)(src + (size_t)row * H + k0 + 16);
        v16bf a = cat8(lo, hi);
        v16bf b = *(const v16bf*)(WlB + (((size_t)kb * 64 + ntile) << 9) + lofs);
        acc = wmma_bf16(a, b, acc);
      }
      float*  h1f  = sBf;                         // overwrites hs0 (already consumed)
      bf16_t* h1b  = sBb;
      float*  outT = out + (size_t)t * H;
      #pragma unroll
      for (int g = 0; g < 8; ++g) {
        const int m = mbase + g;
        float v = tanh_f(acc[g]);
        h1f[(size_t)m * H + n]        = v;
        h1b[(size_t)m * H + n]        = (bf16_t)v;
        outT[(size_t)m * (T * H) + n] = v;
      }
      grid_barrier(bar, phase);
    }

    // ping-pong state buffers
    { float*  tf = sAf; sAf = sBf; sBf = tf; }
    { bf16_t* tb = sAb; sAb = sBb; sBb = tb; }
  }
}

// ---------------------------------------------------------------------------
// In-place LayerNorm per row of out[B*T, H]: unbiased std (N-1), eps on std.
// ---------------------------------------------------------------------------
__global__ __launch_bounds__(256)
void layernorm_kernel(float* __restrict__ out, const float* __restrict__ a2,
                      const float* __restrict__ b2) {
  __shared__ float red[256];
  const int tid = threadIdx.x;
  float* row = out + (size_t)blockIdx.x * H;

  float v[4];
  float s = 0.0f;
  #pragma unroll
  for (int i = 0; i < 4; ++i) { v[i] = row[tid + i * 256]; s += v[i]; }
  red[tid] = s; __syncthreads();
  for (int off = 128; off > 0; off >>= 1) {
    if (tid < off) red[tid] += red[tid + off];
    __syncthreads();
  }
  float mean = red[0] * (1.0f / H);
  __syncthreads();

  float s2 = 0.0f;
  #pragma unroll
  for (int i = 0; i < 4; ++i) { float d = v[i] - mean; s2 += d * d; }
  red[tid] = s2; __syncthreads();
  for (int off = 128; off > 0; off >>= 1) {
    if (tid < off) red[tid] += red[tid + off];
    __syncthreads();
  }
  float stdv = sqrtf(red[0] * (1.0f / (H - 1)));
  float inv  = frcp(stdv + EPS);
  #pragma unroll
  for (int i = 0; i < 4; ++i) {
    int idx = tid + i * 256;
    row[idx] = a2[idx] * (v[i] - mean) * inv + b2[idx];
  }
}

// ---------------------------------------------------------------------------
extern "C" void kernel_launch(void* const* d_in, const int* in_sizes, int n_in,
                              void* d_out, int out_size, void* d_ws, size_t ws_size,
                              hipStream_t stream) {
  const float* X    = (const float*)d_in[0];
  const float* G    = (const float*)d_in[1];
  // d_in[2] = L : unused by the reference output (attention result discarded)
  const float* Wih  = (const float*)d_in[3];
  const float* Whh  = (const float*)d_in[4];
  const float* Wlin = (const float*)d_in[5];
  const float* a2   = (const float*)d_in[6];
  const float* b2   = (const float*)d_in[7];
  float* out = (float*)d_out;

  // Workspace carve (~131 MB): bf16 weight blobs (L2-resident) + state.
  char* p = (char*)d_ws;
  auto take = [&](size_t bytes) -> char* {
    char* q = p; p += (bytes + 255) & ~(size_t)255; return q;
  };
  bf16_t*  WiB = (bf16_t*)take((size_t)NL * 3 * H * H * 2);
  bf16_t*  WhB = (bf16_t*)take((size_t)NL * 3 * H * H * 2);
  bf16_t*  WlB = (bf16_t*)take((size_t)H * 2 * H * 2);
  float*   hAf = (float*)take((size_t)NL * B * H * 4);
  float*   hBf = (float*)take((size_t)NL * B * H * 4);
  bf16_t*  hAb = (bf16_t*)take((size_t)NL * B * H * 2);
  bf16_t*  hBb = (bf16_t*)take((size_t)NL * B * H * 2);
  GridBar* bar = (GridBar*)take(256);

  const size_t wtot = (size_t)2 * NL * 3 * H * H + (size_t)H * 2 * H;
  conv_weights_kernel<<<(unsigned)(wtot / 256), 256, 0, stream>>>(Wih, Whh, Wlin,
                                                                  WiB, WhB, WlB);
  conv_state_kernel<<<(NL * B * H) / 256, 256, 0, stream>>>(G, hAf, hAb, bar);

  fused_decoder_kernel<<<GRID_BLOCKS, 256, 0, stream>>>(X, WiB, WhB, WlB,
                                                        hAf, hBf, hAb, hBb,
                                                        out, bar);

  layernorm_kernel<<<B * T, 256, 0, stream>>>(out, a2, b2);

  (void)in_sizes; (void)n_in; (void)out_size; (void)ws_size;
}
